// EntropyGuidedAttention_76184129896929
// MI455X (gfx1250) — matrix operand order repaired and checked by reading, not compiled, not run on hardware
//
#include <hip/hip_runtime.h>
#include <hip/hip_bf16.h>
#include <math.h>

#define DEV_INLINE __device__ __forceinline__

typedef __bf16 bf16;
typedef __attribute__((ext_vector_type(16))) __bf16 v16bf;
typedef __attribute__((ext_vector_type(8)))  float  v8f;

constexpr int kB = 16, kD = 768, kN = 4096, kQ = 128;
constexpr int TM   = 32;        // token rows per workgroup in fused kernel
constexpr int LDQ  = kD + 16;   // 784: LDS row stride (bf16) for sVF/sQ (16B-aligned rows)
constexpr int LDSP = kQ + 8;    // 136: LDS row stride for S (f32) / P (bf16)

// ---------- helpers ----------

union FragAB { v16bf v; uint4 q[2]; bf16 s[16]; };
union PairBF { unsigned u; bf16 h[2]; };

// A fragment (16x32 bf16, row-major source). Per ISA layout:
// lanes 0-15: M=lane, K = {0..7, 16..23}; lanes 16-31: M=lane-16, K = {8..15, 24..31}
DEV_INLINE v16bf load_a(const bf16* p, int ld, int m0, int kk, int lane) {
  int m  = m0 + (lane & 15);
  int kh = (lane >> 4) * 8;
  FragAB f;
  f.q[0] = *(const uint4*)(p + (size_t)m * ld + kk + kh);
  f.q[1] = *(const uint4*)(p + (size_t)m * ld + kk + 16 + kh);
  return f.v;
}

// B fragment (32x16 bf16, B[k][n] with column n = row n of a row-major [N x K] source).
// lanes 0-15: N=lane, K = {0..7, 16..23}; lanes 16-31: N=lane-16, K = {8..15, 24..31}
DEV_INLINE v16bf load_b(const bf16* p, int ld, int n0, int kk, int lane) {
  int n  = n0 + (lane & 15);
  int kb = kk + (lane >> 4) * 8;
  FragAB f;
  f.q[0] = *(const uint4*)(p + (size_t)n * ld + kb);
  f.q[1] = *(const uint4*)(p + (size_t)n * ld + kb + 16);
  return f.v;
}

DEV_INLINE v8f wmma_bf16(v16bf a, v16bf b, v8f c) {
  return __builtin_amdgcn_wmma_f32_16x16x32_bf16(false, a, false, b, (short)0, c, false, false);
}

// ---------- prep kernels ----------

__global__ __launch_bounds__(256) void wcvt_kernel(
    const float* wq, const float* wk, const float* wv,
    bf16* wqb, bf16* wkb, bf16* wvb) {
  int i = blockIdx.x * 256 + threadIdx.x;
  if (i < kD * kD) {
    wqb[i] = (bf16)wq[i];
    wkb[i] = (bf16)wk[i];
    wvb[i] = (bf16)wv[i];
  }
}

// per-token entropy of softmax over D, visual tokens (strided but lane-coalesced reads)
__global__ __launch_bounds__(256) void vis_ent_kernel(const float* vis, float* ent) {
  int g = blockIdx.x * 256 + threadIdx.x;       // token id in [0, B*N)
  int b = g / kN, n = g % kN;
  const float* base = vis + (size_t)b * kD * kN + n;
  float m = -INFINITY, s = 0.f, t = 0.f;        // online logsumexp + E[x]
  for (int d = 0; d < kD; ++d) {
    float x = base[(size_t)d * kN];
    if (x > m) { float c = __expf(m - x); s = s * c + 1.f; t = t * c + x; m = x; }
    else       { float e = __expf(x - m); s += e; t += x * e; }
  }
  ent[g] = (m + __logf(s)) - t / s;             // H = logZ - E_p[x]
}

// per-token entropy, text tokens: one wave per row
__global__ __launch_bounds__(256) void txt_ent_kernel(const float* txt, float* ent) {
  int lane = threadIdx.x & 31, wid = threadIdx.x >> 5;
  int tok = blockIdx.x * 8 + wid;               // < B*Q = 2048
  const float* row = txt + (size_t)tok * kD;
  float m = -INFINITY, s = 0.f, t = 0.f;
  for (int j = lane; j < kD; j += 32) {
    float x = row[j];
    if (x > m) { float c = __expf(m - x); s = s * c + 1.f; t = t * c + x; m = x; }
    else       { float e = __expf(x - m); s += e; t += x * e; }
  }
  for (int o = 16; o > 0; o >>= 1) {
    float m2 = __shfl_xor(m, o, 32), s2 = __shfl_xor(s, o, 32), t2 = __shfl_xor(t, o, 32);
    float M = fmaxf(m, m2), c1 = __expf(m - M), c2 = __expf(m2 - M);
    s = s * c1 + s2 * c2; t = t * c1 + t2 * c2; m = M;
  }
  if (lane == 0) ent[tok] = (m + __logf(s)) - t / s;
}

// per-batch softmax over n entries
__global__ void bsm_kernel(const float* e, float* w, int n) {
  __shared__ float red[256];
  __shared__ float sM, sS;
  int tid = threadIdx.x, b = blockIdx.x;
  const float* eb = e + (size_t)b * n;
  float* wb = w + (size_t)b * n;
  float m = -INFINITY;
  for (int i = tid; i < n; i += blockDim.x) m = fmaxf(m, eb[i]);
  red[tid] = m; __syncthreads();
  for (int st = blockDim.x / 2; st > 0; st >>= 1) {
    if (tid < st) red[tid] = fmaxf(red[tid], red[tid + st]);
    __syncthreads();
  }
  if (tid == 0) sM = red[0];
  __syncthreads();
  float s = 0.f;
  for (int i = tid; i < n; i += blockDim.x) s += __expf(eb[i] - sM);
  red[tid] = s; __syncthreads();
  for (int st = blockDim.x / 2; st > 0; st >>= 1) {
    if (tid < st) red[tid] += red[tid + st];
    __syncthreads();
  }
  if (tid == 0) sS = red[0];
  __syncthreads();
  float inv = 1.f / sS;
  for (int i = tid; i < n; i += blockDim.x) wb[i] = __expf(eb[i] - sM) * inv;
}

// ---------- k/v projection (WMMA): k row-major bf16, v transposed [B,D,Q] bf16 ----------

__global__ __launch_bounds__(256) void kvproj_kernel(
    const float* txt, const bf16* wkb, const bf16* wvb,
    const float* bk, const float* bv,
    bf16* kb_out, bf16* vt_out) {
  int lane = threadIdx.x & 31, wid = threadIdx.x >> 5;
  int t  = blockIdx.x * 8 + wid;                // 6144 tiles total
  int mt = t % (kB * kQ / 16);                  // 128 row tiles
  int nt = t / (kB * kQ / 16);                  // 48 col tiles
  int m0 = mt * 16, n0 = nt * 16;
  v8f accK = {}; v8f accV = {};
  int m  = m0 + (lane & 15);
  int kh = (lane >> 4) * 8;
  const float* rowp = txt + (size_t)m * kD;
  for (int kk = 0; kk < kD; kk += 32) {
    FragAB fa;
    const float* c0 = rowp + kk + kh;
    const float* c1 = rowp + kk + 16 + kh;
    #pragma unroll
    for (int j = 0; j < 8; ++j) { fa.s[j] = (bf16)c0[j]; fa.s[8 + j] = (bf16)c1[j]; }
    v16bf a  = fa.v;
    v16bf bK = load_b(wkb, kD, n0, kk, lane);
    v16bf bV = load_b(wvb, kD, n0, kk, lane);
    accK = wmma_bf16(a, bK, accK);
    accV = wmma_bf16(a, bV, accV);
  }
  int dcol = n0 + (lane & 15), hi = lane >> 4;
  float bkc = bk[dcol], bvc = bv[dcol];
  #pragma unroll
  for (int r = 0; r < 8; ++r) {                 // k: [B*Q, D] row-major bf16
    int mr = m0 + r + hi * 8;
    kb_out[(size_t)mr * kD + dcol] = (bf16)(accK[r] + bkc);
  }
  int bi = m0 / kQ, q0 = (m0 % kQ) + hi * 8;    // v^T: [B, D, Q]; 8 contiguous q per lane
  FragAB pv;
  #pragma unroll
  for (int r = 0; r < 8; ++r) pv.s[r] = (bf16)(accV[r] + bvc);
  *(uint4*)(vt_out + ((size_t)bi * kD + dcol) * kQ + q0) = pv.q[0];
}

// ---------- fused q-proj + attention (WMMA) ----------

__global__ __launch_bounds__(256) void fused_attn_kernel(
    const float* vis, const bf16* wqb, const float* bq,
    const bf16* kbuf, const bf16* vtbuf,
    const float* ve, const float* te, float* out) {
  __shared__ __align__(16) bf16 sQ[TM * LDQ];   // 50176 B
  __shared__ __align__(16) bf16 sU[TM * LDQ];   // 50176 B: VF, later S|P
  __shared__ float sTE[kQ];
  __shared__ float sVE[TM];
  __shared__ float sOT[8][16 * 17];             // per-wave transpose staging
  float* sS = (float*)sU;                       // [TM][LDSP] f32
  bf16*  sP = sU + TM * LDSP * 2;               // [TM][LDSP] bf16, after sS region

  int tid = threadIdx.x, lane = tid & 31, wid = tid >> 5;
  int bi  = blockIdx.x / (kN / TM);
  int n0g = (blockIdx.x % (kN / TM)) * TM;

  // Phase 0: stage transposed visual tile [n][d] as bf16 (paired b32 LDS stores)
  for (int i = tid; i < TM * kD / 2; i += 256) {
    int d = (i >> 5) * 2, n = i & (TM - 1);
    const float* p = vis + ((size_t)bi * kD + d) * kN + n0g + n;
    PairBF pk;
    pk.h[0] = (bf16)p[0];
    pk.h[1] = (bf16)p[kN];
    *(unsigned*)(sU + n * LDQ + d) = pk.u;
  }
  if (tid < kQ) sTE[tid] = te[bi * kQ + tid];
  if (tid < TM) sVE[tid] = ve[(size_t)bi * kN + n0g + tid];
  __syncthreads();

  // Phase 1: Q = VF @ Wq^T + bq -> sQ (bf16). 96 tiles, 12 per wave.
  for (int it = 0; it < 12; ++it) {
    int t  = it * 8 + wid;
    int mt = t & 1, nt = t >> 1;
    int m0 = mt * 16, n0 = nt * 16;
    v8f acc = {};
    for (int kk = 0; kk < kD; kk += 32) {
      v16bf a = load_a(sU, LDQ, m0, kk, lane);
      v16bf b = load_b(wqb, kD, n0, kk, lane);
      acc = wmma_bf16(a, b, acc);
    }
    int dcol = n0 + (lane & 15), hi = lane >> 4;
    float bqc = bq[dcol];
    #pragma unroll
    for (int r = 0; r < 8; ++r) {
      int mm = m0 + r + hi * 8;
      sQ[mm * LDQ + dcol] = (bf16)(acc[r] + bqc);
    }
  }
  __syncthreads();

  // Phase 2: S = Q @ K^T, scaled + (ve*te) modulated -> sS (f32). 16 tiles, 2/wave.
  const bf16* kbatch = kbuf + (size_t)bi * kQ * kD;
  for (int i = 0; i < 2; ++i) {
    int t  = wid * 2 + i;
    int mt = t & 1, qt = t >> 1;
    int m0 = mt * 16, q0 = qt * 16;
    v8f acc = {};
    for (int kk = 0; kk < kD; kk += 32) {
      v16bf a = load_a(sQ, LDQ, m0, kk, lane);
      v16bf b = load_b(kbatch, kD, q0, kk, lane);
      acc = wmma_bf16(a, b, acc);
    }
    int qc = q0 + (lane & 15), hi = lane >> 4;
    float tec = sTE[qc] * 0.03608439182435161f;  // 1/sqrt(768)
    #pragma unroll
    for (int r = 0; r < 8; ++r) {
      int mm = m0 + r + hi * 8;
      sS[mm * LDSP + qc] = acc[r] * tec * sVE[mm];
    }
  }
  __syncthreads();

  // Phase 3: softmax over Q per row (8 lanes per row) -> sP (bf16)
  {
    int row = tid >> 3, sub = tid & 7;
    const float* base = sS + row * LDSP;
    float m = -INFINITY;
    for (int j = sub; j < kQ; j += 8) m = fmaxf(m, base[j]);
    for (int o = 4; o > 0; o >>= 1) m = fmaxf(m, __shfl_xor(m, o, 8));
    float s = 0.f;
    for (int j = sub; j < kQ; j += 8) s += __expf(base[j] - m);
    for (int o = 4; o > 0; o >>= 1) s += __shfl_xor(s, o, 8);
    float inv = 1.f / s;
    for (int j = sub; j < kQ; j += 8) sP[row * LDSP + j] = (bf16)(__expf(base[j] - m) * inv);
  }
  __syncthreads();

  // Phase 4: O = P @ V; store transposed to out[B,D,N] via per-wave LDS staging
  const bf16* vtb = vtbuf + (size_t)bi * kD * kQ;
  for (int it = 0; it < 12; ++it) {
    int t  = it * 8 + wid;
    int mt = t & 1, nt = t >> 1;
    int m0 = mt * 16, d0 = nt * 16;
    v8f acc = {};
    #pragma unroll
    for (int kk = 0; kk < kQ; kk += 32) {
      v16bf a = load_a(sP, LDSP, m0, kk, lane);
      v16bf b = load_b(vtb, kQ, d0, kk, lane);
      acc = wmma_bf16(a, b, acc);
    }
    int hi = lane >> 4;
    float* o = sOT[wid];
    #pragma unroll
    for (int r = 0; r < 8; ++r) o[(r + hi * 8) * 17 + (lane & 15)] = acc[r];
    #pragma unroll
    for (int j = 0; j < 8; ++j) {
      int dl = j * 2 + hi, ml = lane & 15;
      out[((size_t)bi * kD + d0 + dl) * kN + n0g + m0 + ml] = o[ml * 17 + dl];
    }
  }
}

// ---------- launch ----------

extern "C" void kernel_launch(void* const* d_in, const int* in_sizes, int n_in,
                              void* d_out, int out_size, void* d_ws, size_t ws_size,
                              hipStream_t stream) {
  (void)in_sizes; (void)n_in; (void)out_size; (void)ws_size;
  const float* vis = (const float*)d_in[0];
  const float* txt = (const float*)d_in[1];
  const float* Wq  = (const float*)d_in[2];
  const float* bq  = (const float*)d_in[3];
  const float* Wk  = (const float*)d_in[4];
  const float* bk  = (const float*)d_in[5];
  const float* Wv  = (const float*)d_in[6];
  const float* bv  = (const float*)d_in[7];
  float* out = (float*)d_out;

  char* ws = (char*)d_ws;
  float* entv  = (float*)(ws + 0);              // 262144 B
  float* ve    = (float*)(ws + 262144);         // 262144 B
  float* entt  = (float*)(ws + 524288);         //   8192 B
  float* te    = (float*)(ws + 532480);         //   8192 B
  bf16*  wqb   = (bf16*)(ws + 540672);          // 1179648 B
  bf16*  wkb   = (bf16*)(ws + 1720320);         // 1179648 B
  bf16*  wvb   = (bf16*)(ws + 2899968);         // 1179648 B
  bf16*  kbuf  = (bf16*)(ws + 4079616);         // 3145728 B
  bf16*  vtbuf = (bf16*)(ws + 7225344);         // 3145728 B

  wcvt_kernel<<<(kD * kD + 255) / 256, 256, 0, stream>>>(Wq, Wk, Wv, wqb, wkb, wvb);
  vis_ent_kernel<<<kB * kN / 256, 256, 0, stream>>>(vis, entv);
  txt_ent_kernel<<<kB * kQ / 8, 256, 0, stream>>>(txt, entt);
  bsm_kernel<<<kB, 256, 0, stream>>>(entv, ve, kN);
  bsm_kernel<<<kB, 128, 0, stream>>>(entt, te, kQ);
  kvproj_kernel<<<(kB * kQ / 16) * (kD / 16) / 8, 256, 0, stream>>>(
      txt, wkb, wvb, bk, bv, kbuf, vtbuf);
  fused_attn_kernel<<<kB * (kN / TM), 256, 0, stream>>>(
      vis, wqb, bq, kbuf, vtbuf, ve, te, out);
}
